// IntraModalMamba_61761629716608
// MI455X (gfx1250) — matrix operand order repaired and checked by reading, not compiled, not run on hardware
//
#include <hip/hip_runtime.h>
#include <hip/hip_bf16.h>

// ---------------------------------------------------------------------------
// Bidirectional Mamba block for MI455X (gfx1250, wave32, WMMA).
// Pipeline per direction (directions run sequentially, sharing scratch):
//   GEMM1 (bf16 WMMA) : xz = xn @ in_w^T        -> xi(bf16), z(bf16)
//   conv4+SiLU        : causal (fwd) / anti-causal (bwd) -> xc(f32), xc(bf16)
//   GEMM2 (bf16 WMMA) : xdbl = xc @ xproj_w^T   -> dtrank(bf16), B|C(f32)
//   GEMM3 (bf16 WMMA) : dt = softplus(dtrank @ dt_w^T + dt_b) -> dt(f32)
//   scan (fused gate) : h recurrence over L, ybar=(y+xc*D)*silu(z) -> bf16
//   GEMM4 (bf16 WMMA) : out (+)= ybar @ out_w^T  (beta=0 fwd, 1 bwd)
// GEMM: 256-thread blocks (8 waves), 128M x (16*NSUB)N tile, B tile staged
// in LDS (shared by all waves), A fragments straight from L2.
// ---------------------------------------------------------------------------

#define D_MODEL  512
#define D_INNER  1024
#define D_STATE  16
#define D_CONV   4
#define DT_RANK  32
#define BATCH    8
#define SEQLEN   4096
#define NTOK     (BATCH * SEQLEN)      // 32768

typedef __bf16 bf16_t;
typedef __attribute__((ext_vector_type(16))) __bf16 v16bf;
typedef __attribute__((ext_vector_type(8)))  __bf16 v8bf;
typedef __attribute__((ext_vector_type(8)))  float  v8f;

// ---------------- fragment loaders (16x16x32 bf16 WMMA) --------------------
// A: 16x32 (MxK), row-major, per ISA table:
//   lanes 0-15 : row M=lane,    elems 0..7 = K k0+0..7,  elems 8..15 = K k0+16..23
//   lanes 16-31: row M=lane-16, elems 0..7 = K k0+8..15, elems 8..15 = K k0+24..31
__device__ __forceinline__ v16bf load_a_frag(const bf16_t* __restrict__ A, int lda,
                                             int m0, int k0, int lane) {
    const int half = lane >> 4, r = lane & 15;
    const bf16_t* p0 = A + (size_t)(m0 + r) * lda + (k0 + half * 8);
    const v8bf lo = *reinterpret_cast<const v8bf*>(p0);
    const v8bf hi = *reinterpret_cast<const v8bf*>(p0 + 16);
    v16bf out;
#pragma unroll
    for (int i = 0; i < 8; ++i) { out[i] = lo[i]; out[8 + i] = hi[i]; }
    return out;
}

// B: 32x16 (KxN); tile staged in LDS as rows of 32 bf16 (row = output col n).
// lane n in 0-15 -> col n, K 0..7,16..23; lanes 16-31 -> col lane-16, K 8..15,24..31.
__device__ __forceinline__ v16bf load_b_frag_lds(const bf16_t* sB /*[rows][32]*/,
                                                 int nloc, int lane) {
    const int half = lane >> 4, r = lane & 15;
    const bf16_t* p = sB + (size_t)(nloc + r) * 32 + half * 8;
    const v8bf lo = *reinterpret_cast<const v8bf*>(p);
    const v8bf hi = *reinterpret_cast<const v8bf*>(p + 16);
    v16bf out;
#pragma unroll
    for (int i = 0; i < 8; ++i) { out[i] = lo[i]; out[8 + i] = hi[i]; }
    return out;
}

// ------------------------------ GEMM -------------------------------------
// Block = 256 threads (8 waves). Wave w owns M-tile m0 = by*128 + w*16; all
// waves share the N-range [bx*16*NSUB, ...) whose B tile lives in LDS.
// Epilogues: 0 = split bf16/bf16 (xz -> xi|z)
//            1 = split bf16/f32  (xdbl -> dtrank | B,C)
//            2 = +bias, softplus, f32 (dt)
//            3 = f32 write or accumulate into d_out
template <int EPI, int NSUB>
__global__ void __launch_bounds__(256)
gemm_wmma_bf16(const bf16_t* __restrict__ A, const bf16_t* __restrict__ W,
               int M, int N, int K,
               float* __restrict__ of32, bf16_t* __restrict__ ob0,
               bf16_t* __restrict__ ob1, const float* __restrict__ bias,
               int split, int beta) {
    constexpr int ROWS = 16 * NSUB;                 // B-tile rows (output cols)
    __shared__ alignas(16) bf16_t sB[ROWS * 32];    // ROWS x 32K bf16

    const int tid  = threadIdx.x;
    const int lane = tid & 31;
    const int wave = tid >> 5;
    const int m0 = blockIdx.y * 128 + wave * 16;
    const int n0 = blockIdx.x * ROWS;

    v8f acc[NSUB];
#pragma unroll
    for (int j = 0; j < NSUB; ++j)
#pragma unroll
        for (int e = 0; e < 8; ++e) acc[j][e] = 0.0f;

    for (int k0 = 0; k0 < K; k0 += 32) {
        __syncthreads();   // previous tile fully consumed
        // cooperative B-tile load: ROWS rows x 64B, 16B chunks round-robin
#pragma unroll
        for (int c = tid; c < ROWS * 4; c += 256) {
            const int row = c >> 2, ch = c & 3;
            *reinterpret_cast<v8bf*>(&sB[row * 32 + ch * 8]) =
                *reinterpret_cast<const v8bf*>(W + (size_t)(n0 + row) * K + k0 + ch * 8);
        }
        __syncthreads();

        const v16bf af = load_a_frag(A, K, m0, k0, lane);
#pragma unroll
        for (int j = 0; j < NSUB; ++j) {
            const v16bf bfr = load_b_frag_lds(sB, j * 16, lane);
            acc[j] = __builtin_amdgcn_wmma_f32_16x16x32_bf16(
                false, af, false, bfr, (short)0, acc[j], false, false);
        }
    }

    const int half = lane >> 4, r = lane & 15;
#pragma unroll
    for (int j = 0; j < NSUB; ++j) {
        const int col = n0 + j * 16 + r;
#pragma unroll
        for (int e = 0; e < 8; ++e) {
            const int m = m0 + half * 8 + e;
            const float v = acc[j][e];
            if constexpr (EPI == 0) {
                if (col < split) ob0[(size_t)m * split + col] = (bf16_t)v;
                else             ob1[(size_t)m * split + (col - split)] = (bf16_t)v;
            } else if constexpr (EPI == 1) {
                if (col < split) ob0[(size_t)m * split + col] = (bf16_t)v;
                else             of32[(size_t)m * split + (col - split)] = v;
            } else if constexpr (EPI == 2) {
                const float t = v + bias[col];
                of32[(size_t)m * N + col] = (t > 20.0f) ? t : log1pf(__expf(t));
            } else {
                float* po = &of32[(size_t)m * N + col];
                *po = beta ? (*po + v) : v;
            }
        }
    }
}

// --------------------------- LayerNorm ------------------------------------
// One wave per token (512 elems = 16 per lane), wave32 shuffle reduction.
__global__ void __launch_bounds__(256)
ln_kernel(const float* __restrict__ x, const float* __restrict__ gamma,
          const float* __restrict__ beta, bf16_t* __restrict__ xn) {
    const int wid  = (blockIdx.x * blockDim.x + threadIdx.x) >> 5;
    const int lane = threadIdx.x & 31;
    if (wid >= NTOK) return;
    const float* xr = x + (size_t)wid * D_MODEL;
    float v[16], s = 0.0f, s2 = 0.0f;
#pragma unroll
    for (int i = 0; i < 16; ++i) {
        v[i] = xr[lane + i * 32];
        s += v[i];
        s2 += v[i] * v[i];
    }
#pragma unroll
    for (int off = 16; off >= 1; off >>= 1) {
        s  += __shfl_xor(s,  off, 32);
        s2 += __shfl_xor(s2, off, 32);
    }
    const float mean = s * (1.0f / D_MODEL);
    const float var  = s2 * (1.0f / D_MODEL) - mean * mean;
    const float inv  = rsqrtf(var + 1e-5f);
    bf16_t* orow = xn + (size_t)wid * D_MODEL;
#pragma unroll
    for (int i = 0; i < 16; ++i) {
        const int c = lane + i * 32;
        orow[c] = (bf16_t)(((v[i] - mean) * inv) * gamma[c] + beta[c]);
    }
}

// ---------------- depthwise conv4 + SiLU (causal / anti-causal) ------------
__global__ void __launch_bounds__(256)
conv_silu_kernel(const bf16_t* __restrict__ xi, const float* __restrict__ cw,
                 const float* __restrict__ cb, float* __restrict__ xc,
                 bf16_t* __restrict__ xcb, int rev) {
    const long long idx = (long long)blockIdx.x * blockDim.x + threadIdx.x;
    if (idx >= (long long)NTOK * D_INNER) return;
    const int d = (int)(idx & (D_INNER - 1));
    const long long bl = idx >> 10;
    const int l = (int)(bl & (SEQLEN - 1));
    const int b = (int)(bl >> 12);
    float acc = cb[d];
    const long long base = ((long long)b << 12);
    if (!rev) {  // causal: taps l-3..l, weights w[0..3]
#pragma unroll
        for (int k = 0; k < D_CONV; ++k) {
            const int p = l - 3 + k;
            if (p >= 0)
                acc += (float)xi[((base + p) << 10) + d] * cw[d * D_CONV + k];
        }
    } else {     // anti-causal: taps l..l+3, weights w[3..0]
#pragma unroll
        for (int j = 0; j < D_CONV; ++j) {
            const int p = l + j;
            if (p < SEQLEN)
                acc += (float)xi[((base + p) << 10) + d] * cw[d * D_CONV + (3 - j)];
        }
    }
    const float s = acc / (1.0f + __expf(-acc));
    xc[idx]  = s;
    xcb[idx] = (bf16_t)s;
}

// ----------------------- selective scan + gating ---------------------------
// Block = 256 threads = 256 channels of one batch; grid = (D_INNER/256, B).
// B_t/C_t (32 floats per step) broadcast through LDS.
__global__ void __launch_bounds__(256)
scan_kernel(const float* __restrict__ dt, const float* __restrict__ bc,
            const float* __restrict__ xc, const bf16_t* __restrict__ z,
            const float* __restrict__ A, const float* __restrict__ Dp,
            bf16_t* __restrict__ ybar, int rev) {
    const int tid = threadIdx.x;
    const int d = blockIdx.x * 256 + tid;
    const int b = blockIdx.y;
    float Areg[D_STATE], h[D_STATE];
#pragma unroll
    for (int s = 0; s < D_STATE; ++s) {
        Areg[s] = A[(size_t)d * D_STATE + s];
        h[s] = 0.0f;
    }
    const float Dd = Dp[d];
    __shared__ float sBC[2 * D_STATE];

    for (int step = 0; step < SEQLEN; ++step) {
        const int l = rev ? (SEQLEN - 1 - step) : step;
        const size_t tok = (size_t)b * SEQLEN + l;
        if (tid < 2 * D_STATE) sBC[tid] = bc[tok * (2 * D_STATE) + tid];
        __syncthreads();
        const float dtv = dt[tok * D_INNER + d];
        const float xv  = xc[tok * D_INNER + d];
        const float zv  = (float)z[tok * D_INNER + d];
        const float dtx = dtv * xv;
        float y = 0.0f;
#pragma unroll
        for (int s = 0; s < D_STATE; ++s) {
            const float e = __expf(Areg[s] * dtv);
            h[s] = h[s] * e + dtx * sBC[s];
            y += h[s] * sBC[D_STATE + s];
        }
        const float sil = zv / (1.0f + __expf(-zv));
        ybar[tok * D_INNER + d] = (bf16_t)((y + xv * Dd) * sil);
        __syncthreads();
    }
}

// ------------------------------ prep --------------------------------------
__global__ void cvt_bf16_kernel(const float* __restrict__ in,
                                bf16_t* __restrict__ out, int n) {
    const int i = blockIdx.x * blockDim.x + threadIdx.x;
    if (i < n) out[i] = (bf16_t)in[i];
}
__global__ void negexp_kernel(const float* __restrict__ in,
                              float* __restrict__ out, int n) {
    const int i = blockIdx.x * blockDim.x + threadIdx.x;
    if (i < n) out[i] = -__expf(in[i]);
}

// ---------------------------------------------------------------------------
extern "C" void kernel_launch(void* const* d_in, const int* in_sizes, int n_in,
                              void* d_out, int out_size, void* d_ws, size_t ws_size,
                              hipStream_t stream) {
    (void)in_sizes; (void)n_in; (void)out_size; (void)ws_size;
    const float* x     = (const float*)d_in[0];
    const float* gamma = (const float*)d_in[1];
    const float* beta  = (const float*)d_in[2];
    float* out = (float*)d_out;

    // ---- workspace carve-up (aligned 256B) ----
    char* base = (char*)d_ws;
    size_t off = 0;
    auto carve = [&](size_t bytes) {
        void* p = base + off;
        off = (off + bytes + 255) & ~(size_t)255;
        return p;
    };
    bf16_t* xn_bf   = (bf16_t*)carve((size_t)NTOK * D_MODEL * 2);      // 32 MB
    bf16_t* xi_bf   = (bf16_t*)carve((size_t)NTOK * D_INNER * 2);      // 64 MB
    bf16_t* z_bf    = (bf16_t*)carve((size_t)NTOK * D_INNER * 2);      // 64 MB
    float*  xc_f32  = (float*) carve((size_t)NTOK * D_INNER * 4);      // 128 MB
    bf16_t* xc_bf   = (bf16_t*)carve((size_t)NTOK * D_INNER * 2);      // 64 MB (reused as ybar)
    bf16_t* dtr_bf  = (bf16_t*)carve((size_t)NTOK * DT_RANK * 2);      // 2 MB
    float*  bc_f32  = (float*) carve((size_t)NTOK * 2 * D_STATE * 4);  // 4 MB
    float*  dt_f32  = (float*) carve((size_t)NTOK * D_INNER * 4);      // 128 MB
    bf16_t* w_in    = (bf16_t*)carve((size_t)2 * D_INNER * D_MODEL * 2);
    bf16_t* w_xp    = (bf16_t*)carve((size_t)(DT_RANK + 2 * D_STATE) * D_INNER * 2);
    bf16_t* w_dt    = (bf16_t*)carve((size_t)D_INNER * DT_RANK * 2);
    bf16_t* w_out   = (bf16_t*)carve((size_t)D_MODEL * D_INNER * 2);
    float*  A_f32   = (float*) carve((size_t)D_INNER * D_STATE * 4);

    // ---- LayerNorm (once) ----
    ln_kernel<<<NTOK / 8, 256, 0, stream>>>(x, gamma, beta, xn_bf);

    for (int dir = 0; dir < 2; ++dir) {
        const int i0 = 3 + dir * 9;
        const float* in_w    = (const float*)d_in[i0 + 0];
        const float* conv_w  = (const float*)d_in[i0 + 1];
        const float* conv_b  = (const float*)d_in[i0 + 2];
        const float* xproj_w = (const float*)d_in[i0 + 3];
        const float* dt_w    = (const float*)d_in[i0 + 4];
        const float* dt_b    = (const float*)d_in[i0 + 5];
        const float* A_log   = (const float*)d_in[i0 + 6];
        const float* Dp      = (const float*)d_in[i0 + 7];
        const float* out_w   = (const float*)d_in[i0 + 8];

        // weight conversion + A = -exp(A_log)
        auto cvt = [&](const float* src, bf16_t* dst, int n) {
            cvt_bf16_kernel<<<(n + 255) / 256, 256, 0, stream>>>(src, dst, n);
        };
        cvt(in_w,    w_in,  2 * D_INNER * D_MODEL);
        cvt(xproj_w, w_xp,  (DT_RANK + 2 * D_STATE) * D_INNER);
        cvt(dt_w,    w_dt,  D_INNER * DT_RANK);
        cvt(out_w,   w_out, D_MODEL * D_INNER);
        negexp_kernel<<<(D_INNER * D_STATE + 255) / 256, 256, 0, stream>>>(
            A_log, A_f32, D_INNER * D_STATE);

        // GEMM1: xz = xn @ in_w^T  (M=32768, N=2048, K=512) -> xi|z (bf16)
        gemm_wmma_bf16<0, 8><<<dim3(2 * D_INNER / 128, NTOK / 128), 256, 0, stream>>>(
            xn_bf, w_in, NTOK, 2 * D_INNER, D_MODEL,
            nullptr, xi_bf, z_bf, nullptr, D_INNER, 0);

        // conv4 + SiLU
        conv_silu_kernel<<<(NTOK * D_INNER) / 256, 256, 0, stream>>>(
            xi_bf, conv_w, conv_b, xc_f32, xc_bf, dir);

        // GEMM2: xdbl = xc @ xproj_w^T  (N=64, K=1024) -> dtrank(bf16) | B,C(f32)
        gemm_wmma_bf16<1, 4><<<dim3(1, NTOK / 128), 256, 0, stream>>>(
            xc_bf, w_xp, NTOK, DT_RANK + 2 * D_STATE, D_INNER,
            bc_f32, dtr_bf, nullptr, nullptr, DT_RANK, 0);

        // GEMM3: dt = softplus(dtrank @ dt_w^T + dt_b)  (N=1024, K=32)
        gemm_wmma_bf16<2, 8><<<dim3(D_INNER / 128, NTOK / 128), 256, 0, stream>>>(
            dtr_bf, w_dt, NTOK, D_INNER, DT_RANK,
            dt_f32, nullptr, nullptr, dt_b, 0, 0);

        // selective scan + gating -> ybar (reuses xc_bf buffer)
        scan_kernel<<<dim3(D_INNER / 256, BATCH), 256, 0, stream>>>(
            dt_f32, bc_f32, xc_f32, z_bf, A_f32, Dp, xc_bf, dir);

        // GEMM4: out (+)= ybar @ out_w^T  (N=512, K=1024)
        gemm_wmma_bf16<3, 8><<<dim3(D_MODEL / 128, NTOK / 128), 256, 0, stream>>>(
            xc_bf, w_out, NTOK, D_MODEL, D_INNER,
            out, nullptr, nullptr, nullptr, 0, dir);
    }
}
